// SelfCompressingRGCNAutoEncoder_31318901522602
// MI455X (gfx1250) — matrix operand order repaired and checked by reading, not compiled, not run on hardware
//
#include <hip/hip_runtime.h>
#include <hip/hip_bf16.h>

// ---------------------------------------------------------------------------
// SelfCompressingRGCNAutoEncoder loss on MI455X (gfx1250).
//
// logit_e = src_e^T * W[et_e] * dst_e + bias[et_e]
// out = mean(softplus(-pos_logits)) + mean(softplus(neg_logits))
//
// Strategy: per 16-edge tile, compute H_t = W[r][rows 16t..16t+15,:] @ DST^T
// with v_wmma_f32_16x16x32_f16 (K=64 -> 2 chunks), for all 8 relations, and
// reduce per-lane against src using the 16x16 f32 C-layout (lane = column e,
// VGPR v = row). Select the relation matching edge_type with a cndmask.
// W (8*64*64 f16 = 64KB) staged into LDS via the Tensor Data Mover.
// ---------------------------------------------------------------------------

typedef _Float16 v8h  __attribute__((ext_vector_type(8)));
typedef _Float16 v16h __attribute__((ext_vector_type(16)));
typedef float    v8f  __attribute__((ext_vector_type(8)));
typedef unsigned int u32x4 __attribute__((ext_vector_type(4)));
typedef int      i32x4 __attribute__((ext_vector_type(4)));
typedef int      i32x8 __attribute__((ext_vector_type(8)));

#define N_NODES     50000
#define N_CLUSTERS  64
#define N_RELATIONS 8
#define N_EDGES     1000000

#define WS_W_OFF 0                      // 8*64*64 f16 = 65536 B
#define WS_A_OFF 65536                  // 50000*64 f16 = 6400000 B
#define WS_P_OFF (65536 + 6400000)      // partials (float)

#define EDGE_BLOCKS 2048
#define WAVES_PER_BLOCK 8
#define GROUPS_TOTAL (2 * N_EDGES / 32) // 62500 (32 edges per wave-group)
#define GROUPS_PER_LIST (N_EDGES / 32)  // 31250

#if __has_builtin(__builtin_amdgcn_tensor_load_to_lds)
#define HAVE_TDM 1
#endif

__device__ __forceinline__ float sigmoidf_(float x) {
  return 1.0f / (1.0f + expf(-x));
}

// 16 contiguous f16 (two b128 loads)
__device__ __forceinline__ v16h load16c(const _Float16* p) {
  v8h lo = *(const v8h*)p;
  v8h hi = *(const v8h*)(p + 8);
  return __builtin_shufflevector(lo, hi, 0,1,2,3,4,5,6,7,8,9,10,11,12,13,14,15);
}
// two 8-f16 runs (A-matrix layout: lane holds K runs {0..7} and {16..23} (+8*half))
__device__ __forceinline__ v16h load2runs(const _Float16* p0, const _Float16* p1) {
  v8h lo = *(const v8h*)p0;
  v8h hi = *(const v8h*)p1;
  return __builtin_shufflevector(lo, hi, 0,1,2,3,4,5,6,7,8,9,10,11,12,13,14,15);
}

// --------------------------- prep kernels ----------------------------------

__global__ __launch_bounds__(256) void prep_weights_kernel(
    const float* __restrict__ icl, const float* __restrict__ log_alpha,
    _Float16* __restrict__ wF16) {
  int i = blockIdx.x * 256 + threadIdx.x;   // 8*64*64 = 32768 exact
  float gate = sigmoidf_(log_alpha[i]) * 1.2f - 0.1f;   // (b-a)=1.2, a=-0.1
  gate = fminf(fmaxf(gate, 0.0f), 1.0f);
  wF16[i] = (_Float16)(sigmoidf_(icl[i]) * gate);
}

__global__ __launch_bounds__(256) void prep_assign_kernel(
    const float* __restrict__ a, _Float16* __restrict__ aF16, int n) {
  int i = blockIdx.x * 256 + threadIdx.x;
  if (i < n) aF16[i] = (_Float16)a[i];
}

// --------------------------- main edge kernel ------------------------------

__global__ __launch_bounds__(256) void edge_kernel(
    const _Float16* __restrict__ wF16,
    const _Float16* __restrict__ aF16,
    const float* __restrict__ absent_bias,
    const int* __restrict__ pos_ei, const int* __restrict__ pos_et,
    const int* __restrict__ neg_ei, const int* __restrict__ neg_et,
    float* __restrict__ partials) {
  __shared__ _Float16 ldsW[N_RELATIONS * 64 * 64];  // 64 KB
  __shared__ float waveAcc[WAVES_PER_BLOCK];

  // ---- stage W into LDS ----
#ifdef HAVE_TDM
  if (threadIdx.x < 32) {  // wave 0 issues one 64KB TDM transfer
    unsigned int lds_off = (unsigned int)(unsigned long long)(size_t)&ldsW[0];
    unsigned long long ga = (unsigned long long)(size_t)wF16;
    // D# group0: count=1, lds_addr, global_addr[56:0], type=2
    u32x4 g0 = { 1u, lds_off,
                 (unsigned int)(ga & 0xFFFFFFFFu),
                 (unsigned int)((ga >> 32) & 0x1FFFFFFu) | (2u << 30) };
    // D# group1: data_size=2B; 1-D tensor: dim0=32768, dim1=1, tile_dim0=32768,
    // tile_dim1=1, stride0=32768.
    i32x8 g1 = { 0x00010000, (int)0x80000000, 0x00010000, (int)0x80000000,
                 1, 32768, 0, 0 };
    i32x4 gz4 = { 0, 0, 0, 0 };
    i32x8 gz8 = { 0, 0, 0, 0, 0, 0, 0, 0 };
    __builtin_amdgcn_tensor_load_to_lds(g0, g1, gz4, gz4, gz8, 0);
    __builtin_amdgcn_s_wait_tensorcnt(0);
  }
#else
  for (int i = threadIdx.x; i < N_RELATIONS * 64 * 64; i += 256)
    ldsW[i] = wF16[i];
#endif
  __syncthreads();

  const int wave = threadIdx.x >> 5;
  const int lane = threadIdx.x & 31;
  const int e    = lane & 15;   // edge within 16-edge subtile (also A row M)
  const int hv   = lane >> 4;   // half: selects K-run / C-row-half

  float waveLoss = 0.0f;

  for (int g = blockIdx.x * WAVES_PER_BLOCK + wave; g < GROUPS_TOTAL;
       g += gridDim.x * WAVES_PER_BLOCK) {
    const bool isPos = (g < GROUPS_PER_LIST);
    const int gg = isPos ? g : (g - GROUPS_PER_LIST);
    const int* __restrict__ ei = isPos ? pos_ei : neg_ei;
    const int* __restrict__ et = isPos ? pos_et : neg_et;

    int etype[2];
    v16h B0[2], B1[2];   // B operand: dst^T chunks (K=0..31, K=32..63)
    v8h  S[2][4];        // src slices per row-tile, matching C layout rows
    #pragma unroll
    for (int s = 0; s < 2; ++s) {
      const int idx = gg * 32 + s * 16 + e;
      const int srcIdx = ei[idx];
      const int dstIdx = ei[N_EDGES + idx];
      etype[s] = et[idx];
      const _Float16* dp = aF16 + (size_t)dstIdx * 64;
      B0[s] = load2runs(dp + 8 * hv, dp + 16 + 8 * hv);        // K = 8*hv..+7, 16+8*hv..+7
      B1[s] = load2runs(dp + 32 + 8 * hv, dp + 48 + 8 * hv);   // K = 32+8*hv..+7, 48+8*hv..+7
      const _Float16* sp = aF16 + (size_t)srcIdx * 64 + 8 * hv;
      #pragma unroll
      for (int t = 0; t < 4; ++t) S[s][t] = *(const v8h*)(sp + 16 * t);
    }

    float sel[2] = {0.0f, 0.0f};
    for (int r = 0; r < N_RELATIONS; ++r) {
      float acc0 = 0.0f, acc1 = 0.0f;
      #pragma unroll
      for (int t = 0; t < 4; ++t) {
        // A = W[r] rows 16t..16t+15; lane row = e; K runs per A-layout.
        const _Float16* wp = ldsW + r * 4096 + (16 * t + e) * 64 + 8 * hv;
        v16h A0 = load2runs(wp,      wp + 16);   // K chunk 0..31
        v16h A1 = load2runs(wp + 32, wp + 48);   // K chunk 32..63
        // H_t = W_tile @ DST^T  (f32 accumulate), per subtile
        v8f H0 = {};
        H0 = __builtin_amdgcn_wmma_f32_16x16x32_f16(false, A1, false, B1[0],
                                                    (short)0, H0, false, false);
        H0 = __builtin_amdgcn_wmma_f32_16x16x32_f16(false, A0, false, B0[0],
                                                    (short)0, H0, false, false);
        v8f H1 = {};
        H1 = __builtin_amdgcn_wmma_f32_16x16x32_f16(false, A1, false, B1[1],
                                                    (short)0, H1, false, false);
        H1 = __builtin_amdgcn_wmma_f32_16x16x32_f16(false, A0, false, B0[1],
                                                    (short)0, H1, false, false);
        // dot with src along rows j = 16t + v + 8*hv (C-layout rows)
        #pragma unroll
        for (int v = 0; v < 8; ++v) {
          acc0 += H0[v] * (float)S[0][t][v];
          acc1 += H1[v] * (float)S[1][t][v];
        }
      }
      // combine the two row-halves (lanes l and l+16 share edge e)
      float full0 = acc0 + __shfl_xor(acc0, 16, 32);
      float full1 = acc1 + __shfl_xor(acc1, 16, 32);
      sel[0] = (etype[0] == r) ? full0 : sel[0];
      sel[1] = (etype[1] == r) ? full1 : sel[1];
    }

    #pragma unroll
    for (int s = 0; s < 2; ++s) {
      float logit = sel[s] + absent_bias[etype[s]];
      float x = isPos ? -logit : logit;
      // stable softplus
      float loss = fmaxf(x, 0.0f) + log1pf(expf(-fabsf(x)));
      waveLoss += (lane < 16) ? loss : 0.0f;  // lanes 16..31 are duplicates
    }
  }

  // wave reduce (wave32)
  #pragma unroll
  for (int o = 16; o >= 1; o >>= 1) waveLoss += __shfl_xor(waveLoss, o, 32);
  if (lane == 0) waveAcc[wave] = waveLoss;
  __syncthreads();
  if (threadIdx.x == 0) {
    float t = 0.0f;
    #pragma unroll
    for (int i = 0; i < WAVES_PER_BLOCK; ++i) t += waveAcc[i];
    partials[blockIdx.x] = t * (1.0f / (float)N_EDGES);
  }
}

// --------------------------- final reduction -------------------------------

__global__ __launch_bounds__(256) void reduce_kernel(
    const float* __restrict__ partials, int n, float* __restrict__ out) {
  __shared__ float sm[256];
  float s = 0.0f;
  for (int i = threadIdx.x; i < n; i += 256) s += partials[i];
  sm[threadIdx.x] = s;
  __syncthreads();
  for (int o = 128; o > 0; o >>= 1) {
    if (threadIdx.x < o) sm[threadIdx.x] += sm[threadIdx.x + o];
    __syncthreads();
  }
  if (threadIdx.x == 0) out[0] = sm[0];
}

// --------------------------- launcher --------------------------------------

extern "C" void kernel_launch(void* const* d_in, const int* in_sizes, int n_in,
                              void* d_out, int out_size, void* d_ws, size_t ws_size,
                              hipStream_t stream) {
  (void)in_sizes; (void)n_in; (void)out_size; (void)ws_size;
  const float* assignments = (const float*)d_in[0];
  const float* icl         = (const float*)d_in[1];
  const float* log_alpha   = (const float*)d_in[2];
  const float* absent_bias = (const float*)d_in[3];
  const int*   pos_ei      = (const int*)d_in[4];
  const int*   pos_et      = (const int*)d_in[5];
  const int*   neg_ei      = (const int*)d_in[6];
  const int*   neg_et      = (const int*)d_in[7];

  char* ws = (char*)d_ws;
  _Float16* wF16     = (_Float16*)(ws + WS_W_OFF);
  _Float16* aF16     = (_Float16*)(ws + WS_A_OFF);
  float*    partials = (float*)(ws + WS_P_OFF);

  prep_weights_kernel<<<N_RELATIONS * 64 * 64 / 256, 256, 0, stream>>>(
      icl, log_alpha, wF16);
  const int an = N_NODES * N_CLUSTERS;
  prep_assign_kernel<<<(an + 255) / 256, 256, 0, stream>>>(assignments, aF16, an);
  edge_kernel<<<EDGE_BLOCKS, 256, 0, stream>>>(
      wF16, aF16, absent_bias, pos_ei, pos_et, neg_ei, neg_et, partials);
  reduce_kernel<<<1, 256, 0, stream>>>(partials, EDGE_BLOCKS, (float*)d_out);
}
